// Xerxes2Attention_20993800143156
// MI455X (gfx1250) — matrix operand order repaired and checked by reading, not compiled, not run on hardware
//
#include <hip/hip_runtime.h>

typedef __bf16 bf16;
typedef bf16  v16bf __attribute__((ext_vector_type(16)));
typedef float v8f   __attribute__((ext_vector_type(8)));
typedef unsigned v4u  __attribute__((ext_vector_type(4)));
typedef int      v4i  __attribute__((ext_vector_type(4)));
typedef int      v8i_ __attribute__((ext_vector_type(8)));

#define B_   2
#define S_   2048
#define E_   2048
#define H_   16
#define D_   128
#define M_   (B_ * S_)      // 4096
#define CC   64             // attention chunk (math-equivalent to ref's 128)
#define NCH  (S_ / CC)      // 32 chunks

#define GLOBAL_AS __attribute__((address_space(1)))
#define LDS_AS    __attribute__((address_space(3)))

// Address-space qualified 128-bit vector pointees for the async builtin.
typedef GLOBAL_AS v4i g_v4i;
typedef LDS_AS    v4i l_v4i;

// -------- CDNA5 feature gates (device pass only; host pass falls back) ------
#if defined(__gfx1250__) && __has_builtin(__builtin_amdgcn_global_load_async_to_lds_b128)
#define HAS_ASYNC_LDS 1
#else
#define HAS_ASYNC_LDS 0
#endif

#if defined(__gfx1250__) && __has_builtin(__builtin_amdgcn_tensor_load_to_lds)
#define HAS_TDM 1
#else
#define HAS_TDM 0
#endif

// ---------------------------------------------------------------------------
// WMMA helpers (CDNA5 bf16 16x16x32, fp32 accumulate)
// ---------------------------------------------------------------------------
__device__ __forceinline__ v8f wmma_bf16(v16bf a, v16bf b, v8f c) {
  return __builtin_amdgcn_wmma_f32_16x16x32_bf16(false, a, false, b, (short)0, c,
                                                 false, false);
}

// A fragment: 16x32 bf16, row-major source with element stride `stride`.
// lanes 0-15 hold M=lane, K=0..15; lanes 16-31 K=16..31.
__device__ __forceinline__ v16bf load_frag_a(const bf16* base, int row0, int k0,
                                             int stride) {
  int lane = threadIdx.x & 31;
  int half = lane >> 4;
  const bf16* p = base + (row0 + (lane & 15)) * stride + k0 + 16 * half;
  union { v16bf v; unsigned u[8]; } f;
#pragma unroll
  for (int j = 0; j < 8; ++j) {
    int k = 2 * j;
    f.u[j] = *(const unsigned*)(p + k);
  }
  return f.v;
}

// B fragment from Bt stored row-major as [N][K] (element stride `stride`).
__device__ __forceinline__ v16bf load_frag_b(const bf16* base, int col0, int k0,
                                             int stride) {
  int lane = threadIdx.x & 31;
  const bf16* p = base + (col0 + (lane & 15)) * stride + k0 + 16 * (lane >> 4);
  union { v16bf v; unsigned u[8]; } f;
#pragma unroll
  for (int j = 0; j < 8; ++j) f.u[j] = *(const unsigned*)(p + 2 * j);
  return f.v;
}

// B fragment converted on the fly from an fp32 matrix (kv state).
__device__ __forceinline__ v16bf load_frag_b_f32(const float* base, int col0,
                                                 int k0, int stride) {
  int lane = threadIdx.x & 31;
  const float* p = base + (col0 + (lane & 15)) * stride + k0 + 16 * (lane >> 4);
  union { v16bf v; bf16 e[16]; } f;
#pragma unroll
  for (int j = 0; j < 8; ++j) {
    float2 w = *(const float2*)(p + 2 * j);
    f.e[2 * j]     = (bf16)w.x;
    f.e[2 * j + 1] = (bf16)w.y;
  }
  return f.v;
}

// ---------------------------------------------------------------------------
// Async / TDM staging helpers
// ---------------------------------------------------------------------------
__device__ __forceinline__ void async_copy16(bf16* lds, const bf16* g) {
#if HAS_ASYNC_LDS
  __builtin_amdgcn_global_load_async_to_lds_b128((g_v4i*)g, (l_v4i*)lds, 0, 0);
#else
  *(uint4*)lds = *(const uint4*)g;
#endif
}

__device__ __forceinline__ void wait_async_lds() {
#if HAS_ASYNC_LDS
#if __has_builtin(__builtin_amdgcn_s_wait_asynccnt)
  __builtin_amdgcn_s_wait_asynccnt(0);
#else
  asm volatile("s_wait_asynccnt 0x0" ::: "memory");
#endif
#endif
}

__device__ __forceinline__ void wait_tensor() {
#if HAS_TDM
  __builtin_amdgcn_s_wait_tensorcnt(0);
#endif
}

// TDM: 2D tile load, rows x 32 bf16 per row, into LDS with padded pitch
// 80B (= 32 bf16 + 16B pad) matching the 40-element LDS stride.
__device__ __forceinline__ void tdm_load_tile(bf16* lds_dst, const bf16* gsrc,
                                              int rows, int tensor_w,
                                              long row_stride_elems) {
#if HAS_TDM
  unsigned lds_off = (unsigned)(unsigned long)(LDS_AS char*)lds_dst;
  unsigned long ga = (unsigned long)gsrc;
  v4u g0; v8i_ g1; v4i g2 = {}; v4i g3 = {};
  // D# group 0: count=1 | lds_addr | global_addr | type=2
  g0[0] = 1u;
  g0[1] = lds_off;
  g0[2] = (unsigned)(ga & 0xffffffffu);
  g0[3] = (unsigned)((ga >> 32) & 0x01ffffffu) | (2u << 30);
  // D# group 1: mask=0, data_size=2B, pad_enable, pad_interval=16DW (64B),
  // pad_amount=4DW (16B); tensor_dim0/1, tile_dim0=32, tile_dim1=rows,
  // tensor_dim0_stride = row_stride_elems.
  unsigned td0 = (unsigned)tensor_w;
  unsigned td1 = (unsigned)rows;
  unsigned long st0 = (unsigned long)row_stride_elems;
  g1[0] = (int)((1u << 16) | (1u << 20) | (3u << 22) | (3u << 25));
  g1[1] = (int)((td0 & 0xffffu) << 16);
  g1[2] = (int)((td0 >> 16) | ((td1 & 0xffffu) << 16));
  g1[3] = (int)((td1 >> 16) | (32u << 16));
  g1[4] = (int)(td1 & 0xffffu);
  g1[5] = (int)(st0 & 0xffffffffu);
  g1[6] = (int)((st0 >> 32) & 0xffffu);
  g1[7] = 0;
#if __clang_major__ >= 23
  v8i_ zz = {};
  __builtin_amdgcn_tensor_load_to_lds(g0, g1, g2, g3, zz, 0);
#else
  __builtin_amdgcn_tensor_load_to_lds(g0, g1, g2, g3, 0);
#endif
#else
  (void)lds_dst; (void)gsrc; (void)rows; (void)tensor_w; (void)row_stride_elems;
#endif
}

// ---------------------------------------------------------------------------
// Elementwise fp32 -> bf16 convert
// ---------------------------------------------------------------------------
__global__ void f32_to_bf16_kernel(const float* __restrict__ src,
                                   bf16* __restrict__ dst, size_t n) {
  size_t i = (size_t)blockIdx.x * blockDim.x + threadIdx.x;
  size_t st = (size_t)gridDim.x * blockDim.x;
  for (; i < n; i += st) dst[i] = (bf16)src[i];
}

// ---------------------------------------------------------------------------
// Transpose + convert: src fp32 [K][N] row-major -> dst bf16 [N][K]
// ---------------------------------------------------------------------------
__global__ void transpose_f32_to_bf16(const float* __restrict__ src,
                                      bf16* __restrict__ dst, int K, int N) {
  __shared__ float tile[32][33];
  int k0 = blockIdx.y * 32, n0 = blockIdx.x * 32;
  int tx = threadIdx.x, ty = threadIdx.y;  // 32 x 8
#pragma unroll
  for (int j = 0; j < 32; j += 8)
    tile[ty + j][tx] = src[(size_t)(k0 + ty + j) * N + n0 + tx];
  __syncthreads();
#pragma unroll
  for (int j = 0; j < 32; j += 8)
    dst[(size_t)(n0 + ty + j) * K + k0 + tx] = (bf16)tile[tx][ty + j];
}

// ---------------------------------------------------------------------------
// bf16 WMMA GEMM:  C[M][N] = A[M][K] (row-major bf16) x Bt[N][K] (bf16)
// 128x128 block tile, 8 wave32; double-buffered KC=32 stages:
//   A-tile staged by the Tensor Data Mover (tensor_load_to_lds, TENSORcnt),
//   B-tile staged by GLOBAL_LOAD_ASYNC_TO_LDS_B128 (ASYNCcnt).
// Stage N+1 is issued before computing stage N so DMA overlaps WMMA.
// ---------------------------------------------------------------------------
template <typename OutT>
__global__ void __launch_bounds__(256)
gemm_wmma(const bf16* __restrict__ A, const bf16* __restrict__ Bt,
          OutT* __restrict__ C, int M, int N, int K) {
  __shared__ bf16 As[2][128 * 40];
  __shared__ bf16 Bs[2][128 * 40];

  int tid = threadIdx.x;
  int wave = tid >> 5, lane = tid & 31, half = lane >> 4, l15 = lane & 15;
  int wm = wave & 3, wn = wave >> 2;
  int m0 = blockIdx.y * 128, n0 = blockIdx.x * 128;

  v8f acc[2][4] = {};

  // stage loader: A via TDM (wave 0 issues one descriptor), B via async-LDS
  auto load_stage = [&](int buf, int kt) {
#if HAS_TDM
    if (wave == 0)
      tdm_load_tile(&As[buf][0], A + (size_t)m0 * K + kt, 128, K - kt, K);
#else
    for (int i = tid; i < 512; i += 256) {
      int r = i >> 2, c8 = (i & 3) * 8;
      *(uint4*)&As[buf][r * 40 + c8] =
          *(const uint4*)&A[(size_t)(m0 + r) * K + kt + c8];
    }
#endif
    for (int i = tid; i < 512; i += 256) {
      int r = i >> 2, c8 = (i & 3) * 8;
      async_copy16(&Bs[buf][r * 40 + c8], &Bt[(size_t)(n0 + r) * K + kt + c8]);
    }
  };

  load_stage(0, 0);
  wait_async_lds();
  wait_tensor();
  __syncthreads();

  for (int kt = 0; kt < K; kt += 32) {
    int cur = (kt >> 5) & 1;
    if (kt + 32 < K) load_stage(cur ^ 1, kt + 32);  // prefetch next stage

    v16bf a0 = load_frag_a(&As[cur][0], wm * 32, 0, 40);
    v16bf a1 = load_frag_a(&As[cur][0], wm * 32 + 16, 0, 40);
#pragma unroll
    for (int nt = 0; nt < 4; ++nt) {
      v16bf bb = load_frag_b(&Bs[cur][0], wn * 64 + nt * 16, 0, 40);
      acc[0][nt] = wmma_bf16(a0, bb, acc[0][nt]);
      acc[1][nt] = wmma_bf16(a1, bb, acc[1][nt]);
    }

    wait_async_lds();
    wait_tensor();
    __syncthreads();
  }

#pragma unroll
  for (int mt = 0; mt < 2; ++mt)
#pragma unroll
    for (int nt = 0; nt < 4; ++nt)
#pragma unroll
      for (int i = 0; i < 8; ++i) {
        int m = m0 + wm * 32 + mt * 16 + i + 8 * half;
        int n = n0 + wn * 64 + nt * 16 + l15;
        C[(size_t)m * N + n] = (OutT)acc[mt][nt][i];
      }
}

// ---------------------------------------------------------------------------
// Decay linear attention, chunked scan with fp32 kv state in LDS.
// Grid: B*H*2 blocks (value-dim split in half); 256 threads (8 waves).
// Per chunk (C=64): scores = q k^T (masked decay), out = scores@v + (q*dec)@kv,
// kv = exp(-s*C)*kv + (k*dec)^T @ v  -- all via bf16 WMMA, fp32 accumulate.
// Dynamic LDS = 122880 bytes (uses CDNA5's 320KB/WGP).
// ---------------------------------------------------------------------------
__global__ void __launch_bounds__(256)
attn_kernel(const bf16* __restrict__ Qb, const bf16* __restrict__ Kb,
            const bf16* __restrict__ Vb, const float* __restrict__ amask,
            const float* __restrict__ slopes, float* __restrict__ attn_out) {
  extern __shared__ char smem_raw[];
  float* kvS = (float*)smem_raw;                 // [64 e][132]   (kvS[e][d]=kv[d][e])
  bf16* qb   = (bf16*)(smem_raw + 64 * 132 * 4); // [64 c][136]
  bf16* qdb  = qb + 64 * 136;                    // q * exp(-s*t)
  bf16* kb   = qdb + 64 * 136;                   // [64 c][136]
  bf16* vtb  = kb + 64 * 136;                    // [64 e][72]  v^T (masked)
  bf16* kdtb = vtb + 64 * 72;                    // [128 d][72] (k*kdec)^T
  bf16* smb  = kdtb + 128 * 72;                  // [64 c][72]  masked scores

  int tid = threadIdx.x;
  int wave = tid >> 5, lane = tid & 31, half = lane >> 4, l15 = lane & 15;
  int bid = blockIdx.x;
  int eh = bid & 1, h = (bid >> 1) & (H_ - 1), b = bid >> 5;
  int eo = eh * 64;
  float s = slopes[h];
  float cdec = __expf(-s * (float)CC);

  for (int i = tid; i < 64 * 132; i += 256) kvS[i] = 0.f;
  __syncthreads();

  for (int ci = 0; ci < NCH; ++ci) {
    int c0 = ci * CC;

    // ---- stage chunk operands into LDS (decays / mask / transposes) ----
    for (int i = tid; i < 64 * 16; i += 256) {
      int r = i >> 4, g = (i & 15) * 8;
      size_t off = (size_t)(b * S_ + c0 + r) * E_ + h * D_ + g;
      union { uint4 u; bf16 e[8]; } qv, kv2, qd;
      qv.u = *(const uint4*)(Qb + off);
      kv2.u = *(const uint4*)(Kb + off);
      *(uint4*)(qb + r * 136 + g) = qv.u;
      *(uint4*)(kb + r * 136 + g) = kv2.u;
      float qdec = __expf(-s * (float)r);
      float kdec = __expf(-s * (float)(CC - r));
#pragma unroll
      for (int j = 0; j < 8; ++j) {
        qd.e[j] = (bf16)((float)qv.e[j] * qdec);
        kdtb[(g + j) * 72 + r] = (bf16)((float)kv2.e[j] * kdec);
      }
      *(uint4*)(qdb + r * 136 + g) = qd.u;
    }
    for (int i = tid; i < 64 * 8; i += 256) {
      int r = i >> 3, g = (i & 7) * 8;
      size_t off = (size_t)(b * S_ + c0 + r) * E_ + h * D_ + eo + g;
      union { uint4 u; bf16 e[8]; } vv;
      vv.u = *(const uint4*)(Vb + off);
      float mk = amask[b * S_ + c0 + r];
#pragma unroll
      for (int j = 0; j < 8; ++j)
        vtb[(g + j) * 72 + r] = (bf16)((float)vv.e[j] * mk);
    }
    __syncthreads();

    // ---- scores[c][cj] = q.k^T, masked causal decay, -> bf16 smb ----
#pragma unroll
    for (int t = 0; t < 2; ++t) {
      int id = wave + t * 8, tm = id >> 2, tn = id & 3;
      v8f acc = {};
#pragma unroll
      for (int kk = 0; kk < 4; ++kk) {
        v16bf a  = load_frag_a(qb, tm * 16, kk * 32, 136);
        v16bf bb = load_frag_b(kb, tn * 16, kk * 32, 136);
        acc = wmma_bf16(a, bb, acc);
      }
#pragma unroll
      for (int i = 0; i < 8; ++i) {
        int m = tm * 16 + i + 8 * half, n = tn * 16 + l15;
        int rel = m - n;
        float dv = (rel >= 0) ? acc[i] * __expf(-s * (float)rel) : 0.f;
        smb[m * 72 + n] = (bf16)dv;
      }
    }
    __syncthreads();

    // ---- out[c][e] = smb @ v  +  (q*dec) @ kv ----
#pragma unroll
    for (int t = 0; t < 2; ++t) {
      int id = wave + t * 8, tm = id >> 2, tn = id & 3;
      v8f acc = {};
#pragma unroll
      for (int kk = 0; kk < 2; ++kk) {  // intra, K = 64
        v16bf a  = load_frag_a(smb, tm * 16, kk * 32, 72);
        v16bf bb = load_frag_b(vtb, tn * 16, kk * 32, 72);
        acc = wmma_bf16(a, bb, acc);
      }
#pragma unroll
      for (int kk = 0; kk < 4; ++kk) {  // inter, K = 128, B built from fp32 kv
        v16bf a  = load_frag_a(qdb, tm * 16, kk * 32, 136);
        v16bf bb = load_frag_b_f32(kvS, tn * 16, kk * 32, 132);
        acc = wmma_bf16(a, bb, acc);
      }
#pragma unroll
      for (int i = 0; i < 8; ++i) {
        int m = tm * 16 + i + 8 * half, n = tn * 16 + l15;
        attn_out[(size_t)(b * S_ + c0 + m) * E_ + h * D_ + eo + n] = acc[i];
      }
    }
    __syncthreads();

    // ---- kv state update: kvS = cdec*kvS + v^T @ kd  (M=64 e, N=128 d) ----
#pragma unroll
    for (int t = 0; t < 4; ++t) {
      int id = wave * 4 + t, te = id >> 3, td = id & 7;
      v8f acc;
#pragma unroll
      for (int i = 0; i < 8; ++i) {
        int e = te * 16 + i + 8 * half, d = td * 16 + l15;
        acc[i] = kvS[e * 132 + d] * cdec;
      }
#pragma unroll
      for (int kk = 0; kk < 2; ++kk) {
        v16bf a  = load_frag_a(vtb, te * 16, kk * 32, 72);
        v16bf bb = load_frag_b(kdtb, td * 16, kk * 32, 72);
        acc = wmma_bf16(a, bb, acc);
      }
#pragma unroll
      for (int i = 0; i < 8; ++i) {
        int e = te * 16 + i + 8 * half, d = td * 16 + l15;
        kvS[e * 132 + d] = acc[i];
      }
    }
    __syncthreads();
  }
}

// ---------------------------------------------------------------------------
// RMSNorm over H*D + sigmoid gate; emits bf16 operand for the final GEMM.
// ---------------------------------------------------------------------------
__global__ void __launch_bounds__(256)
rmsgate_kernel(const float* __restrict__ attn, const bf16* __restrict__ Gb,
               const float* __restrict__ normw, bf16* __restrict__ Xg) {
  __shared__ float red[256];
  int row = blockIdx.x, tid = threadIdx.x;
  const float* a = attn + (size_t)row * E_;
  float p = 0.f;
  for (int i = tid; i < E_; i += 256) { float x = a[i]; p += x * x; }
  red[tid] = p;
  __syncthreads();
  for (int off = 128; off > 0; off >>= 1) {
    if (tid < off) red[tid] += red[tid + off];
    __syncthreads();
  }
  float inv = rsqrtf(red[0] / (float)E_ + 1e-6f);
  for (int i = tid; i < E_; i += 256) {
    float g = (float)Gb[(size_t)row * E_ + i];
    float sg = 1.f / (1.f + __expf(-g));
    Xg[(size_t)row * E_ + i] = (bf16)(a[i] * inv * normw[i] * sg);
  }
}

// ---------------------------------------------------------------------------
extern "C" void kernel_launch(void* const* d_in, const int* in_sizes, int n_in,
                              void* d_out, int out_size, void* d_ws,
                              size_t ws_size, hipStream_t stream) {
  (void)in_sizes; (void)n_in; (void)out_size; (void)ws_size;
  const float* hs    = (const float*)d_in[0];
  const float* amask = (const float*)d_in[1];
  const float* slope = (const float*)d_in[2];
  const float* Wsrc[5] = {(const float*)d_in[3], (const float*)d_in[4],
                          (const float*)d_in[5], (const float*)d_in[6],
                          (const float*)d_in[7]};  // Wq Wk Wv Wg Wo
  const float* normw = (const float*)d_in[8];
  float* out = (float*)d_out;

  char* ws = (char*)d_ws;
  size_t o = 0;
  bf16* hsb = (bf16*)(ws + o); o += (size_t)M_ * E_ * 2;   // also reused as Xg
  bf16* WT[5];
  for (int i = 0; i < 5; ++i) { WT[i] = (bf16*)(ws + o); o += (size_t)E_ * E_ * 2; }
  bf16* QK[4];  // Qb, Kb, Vb, Gb (bf16)
  for (int i = 0; i < 4; ++i) { QK[i] = (bf16*)(ws + o); o += (size_t)M_ * E_ * 2; }
  float* attn = (float*)(ws + o); o += (size_t)M_ * E_ * 4;
  bf16* Xg = hsb;  // hs_bf16 no longer needed after the QKVG GEMMs

  // 1) convert hidden_states to bf16
  f32_to_bf16_kernel<<<2048, 256, 0, stream>>>(hs, hsb, (size_t)M_ * E_);

  // 2) transpose+convert all weights to [N][K] bf16
  dim3 tb(32, 8), tg(E_ / 32, E_ / 32);
  for (int i = 0; i < 5; ++i)
    transpose_f32_to_bf16<<<tg, tb, 0, stream>>>(Wsrc[i], WT[i], E_, E_);

  // 3) Q/K/V/G projections (bf16 WMMA, bf16 outputs)
  dim3 gb(256), gg(E_ / 128, M_ / 128);
  for (int i = 0; i < 4; ++i)
    gemm_wmma<bf16><<<gg, gb, 0, stream>>>(hsb, WT[i], QK[i], M_, E_, E_);

  // 4) chunked decay linear attention (big dynamic LDS: 120 KB/WG)
  (void)hipFuncSetAttribute((const void*)attn_kernel,
                            hipFuncAttributeMaxDynamicSharedMemorySize, 122880);
  attn_kernel<<<B_ * H_ * 2, 256, 122880, stream>>>(QK[0], QK[1], QK[2], amask,
                                                    slope, attn);

  // 5) RMSNorm + sigmoid gate -> bf16 operand
  rmsgate_kernel<<<M_, 256, 0, stream>>>(attn, QK[3], normw, Xg);

  // 6) output projection (fp32 result)
  gemm_wmma<float><<<gg, gb, 0, stream>>>(Xg, WT[4], out, M_, E_, E_);
}
